// RGCNNet_39041252720862
// MI455X (gfx1250) — compile-verified, matrix operand order checked by reading, not run on hardware
//
#include <hip/hip_runtime.h>
#include <hip/hip_bf16.h>

typedef __attribute__((ext_vector_type(16))) _Float16 v16h;
typedef __attribute__((ext_vector_type(8)))  float    v8f;

#define D64   64
#define NDIM  6
#define INFEAT 204      // original W column count: 3*64 + 2*6
#define NSTR  72        // node_rep row stride (f16 elems) = 144B, 16B aligned, [70,71]=0
#define KT_   7         // K tiles of 32 -> KPAD = 224
#define ASTR  232       // LDS A-tile row stride (f16): 464B -> 16B-aligned rows for async b128
#define MTILE 64        // edges per block
#define CSTR  68        // LDS C staging row stride (floats)

// K-axis layout (new-k): [0..71]=nrep[row] (incl 2 zero pads), [72..143]=nrep[col],
// [144..207]=ea, [208..223]=0 (zeroed), [224..231] unread.

// ---------------- utility kernels ----------------
__global__ void zero_f32(float* p, long n){
  long i = blockIdx.x*(long)blockDim.x + threadIdx.x;
  if(i<n) p[i]=0.f;
}
__global__ void zero_i32(int* p, int n){
  int i = blockIdx.x*blockDim.x + threadIdx.x;
  if(i<n) p[i]=0;
}
__global__ void count_batch(const int* batch, int* counts, int N){
  int i = blockIdx.x*blockDim.x+threadIdx.x;
  if(i<N) atomicAdd(&counts[batch[i]], 1);
}
__global__ void scan_heads(const int* counts, int* head, int* tail, int B){
  if(blockIdx.x==0 && threadIdx.x==0){
    int acc=0;
    for(int g=0; g<B; ++g){ head[g]=acc; tail[g]=acc+1; acc+=counts[g]; }
  }
}
__global__ void scatter_x0(const float* x, const int* head, const int* tail, float* x0, int B){
  int i = blockIdx.x*blockDim.x+threadIdx.x;
  if(i < B*NDIM){
    int g=i/NDIM, d=i%NDIM;
    int h=head[g], t=tail[g];
    x0[h*NDIM+d] = x[h*NDIM+d];
    x0[t*NDIM+d] = x[t*NDIM+d];
  }
}
__global__ void edge_cnt(const int* col, const float* mask, float* cnt, int E){
  int e=blockIdx.x*blockDim.x+threadIdx.x;
  if(e<E) atomicAdd(&cnt[col[e]], mask[e]);
}
// materialize ea0 = edge_emb[edge_rel] as f16
__global__ void gather_emb(const int* rel, const float* emb, _Float16* ea, int E){
  long i = blockIdx.x*(long)blockDim.x+threadIdx.x;   // one 4-elem chunk per thread
  if(i < (long)E*16){
    int e = (int)(i>>4), q = (int)(i&15);
    float4 v = ((const float4*)emb)[(long)rel[e]*16 + q];
    union { _Float16 h[4]; uint2 u; } pk;
    pk.h[0]=(_Float16)v.x; pk.h[1]=(_Float16)v.y; pk.h[2]=(_Float16)v.z; pk.h[3]=(_Float16)v.w;
    ((uint2*)ea)[i] = pk.u;
  }
}

// Pre-swizzle W into per-lane WMMA B-fragment layout over the NEW K axis.
__global__ void prep_w(const float* W, _Float16* wswz){
  int idx = blockIdx.x*blockDim.x+threadIdx.x;
  if(idx >= KT_*4*32*16) return;
  int j  = idx & 15;
  int l  = (idx>>4)&31;
  int nt = (idx>>9)&3;
  int kt = idx>>11;
  int v = j>>1, khalf = l>>4;
  int klocal = ((v>=4)?16:0) + khalf*8 + (v&3)*2 + (j&1);
  int k = kt*32 + klocal;            // new-k
  int n = nt*16 + (l&15);            // output channel
  int oldk = -1;
  if(k < 70)        oldk = k;
  else if(k < 72)   oldk = -1;
  else if(k < 142)  oldk = k-2;
  else if(k < 144)  oldk = -1;
  else if(k < 208)  oldk = k-4;
  float val = (oldk >= 0) ? W[n*INFEAT + oldk] : 0.f;
  wswz[idx] = (_Float16)val;
}

__global__ void edge_agg_f16(const int* col, const float* mask, const _Float16* ea, float* agg, int E){
  long i = blockIdx.x*(long)blockDim.x+threadIdx.x;   // 4 channels per thread
  if(i < (long)E*16){
    int e=(int)(i>>4), q=(int)(i&15);
    const _Float16* p = ea + (long)e*D64 + q*4;
    float m = mask[e];
    float* dst = &agg[(long)col[e]*D64 + q*4];
    atomicAdd(dst+0, (float)p[0]*m);
    atomicAdd(dst+1, (float)p[1]*m);
    atomicAdd(dst+2, (float)p[2]*m);
    atomicAdd(dst+3, (float)p[3]*m);
  }
}
__global__ void edge_agg_f32(const int* col, const float* mask, const float* ea, float* agg, int E){
  long i = blockIdx.x*(long)blockDim.x+threadIdx.x;
  if(i < (long)E*16){
    int e=(int)(i>>4), q=(int)(i&15);
    float4 v = ((const float4*)ea)[i];
    float m = mask[e];
    float* dst = &agg[(long)col[e]*D64 + q*4];
    atomicAdd(dst+0, v.x*m);
    atomicAdd(dst+1, v.y*m);
    atomicAdd(dst+2, v.z*m);
    atomicAdd(dst+3, v.w*m);
  }
}
// node_rep stored as f16, stride NSTR, pad cols [70,71] = 0
__global__ void node_build(const float* agg, const float* cnt, const float* x0, _Float16* nrep, int N){
  long i = blockIdx.x*(long)blockDim.x+threadIdx.x;
  if(i < (long)N*NSTR){
    int n=(int)(i/NSTR), c=(int)(i%NSTR);
    float v;
    if(c < D64)      v = agg[(long)n*D64+c] / (cnt[n]+1.f);
    else if(c < 70)  v = x0[n*NDIM + (c-D64)];
    else             v = 0.f;
    nrep[i] = (_Float16)v;
  }
}

// -------- fused async-gather + WMMA GEMM + bias + ReLU (64 edges / block) --------
// Gather = pure f16 memcpy -> per-lane GLOBAL_LOAD_ASYNC_TO_LDS_B128 (ASYNCcnt),
// drained with s_wait_asynccnt 0 + workgroup barrier before WMMA.
template<bool OUTF16>
__global__ __launch_bounds__(128) void edge_gemm(
    const int* __restrict__ row, const int* __restrict__ col,
    const _Float16* __restrict__ nrep, const _Float16* __restrict__ ea_in,
    const _Float16* __restrict__ wswz, const float* __restrict__ bias,
    void* __restrict__ ea_out, int E){
  __shared__ __align__(16) char smem[MTILE*ASTR*2];      // 29696 B: A-tile (f16) / C staging (f32)
  __shared__ int rIdx[MTILE], cIdx[MTILE];
  _Float16* Alds = (_Float16*)smem;
  float*    Cst  = (float*)smem;
  const int t = threadIdx.x;
  const long ebase = (long)blockIdx.x * MTILE;

  if(t < MTILE){        long e = ebase + t;        rIdx[t]    = (e < E) ? row[e] : 0; }
  else {                long e = ebase + (t-64);   cIdx[t-64] = (e < E) ? col[e] : 0; }
  __syncthreads();

  // ---- async gather: 64 edges x 26 b128 chunks (9 row-nrep + 9 col-nrep + 8 ea) ----
  for(int item = t; item < MTILE*26; item += 128){
    int s = item/26, ch = item%26;
    long e = ebase + s;
    if(e < E){
      const _Float16* src;
      int fb;
      if(ch < 9)       { src = nrep + (long)rIdx[s]*NSTR + ch*8;       fb = ch*8; }
      else if(ch < 18) { src = nrep + (long)cIdx[s]*NSTR + (ch-9)*8;   fb = 72 + (ch-9)*8; }
      else             { src = ea_in + e*D64 + (ch-18)*8;              fb = 144 + (ch-18)*8; }
      unsigned          lds = (unsigned)(size_t)&Alds[s*ASTR + fb];    // low 32 bits = LDS offset
      unsigned long long ga = (unsigned long long)(size_t)src;
      asm volatile("global_load_async_to_lds_b128 %0, %1, off"
                   :: "v"(lds), "v"(ga) : "memory");
    }
  }
  // zero K-pad [208, ASTR)
  for(int p = t; p < MTILE*(ASTR-208); p += 128){
    int s = p/(ASTR-208), f = 208 + p%(ASTR-208);
    Alds[s*ASTR + f] = (_Float16)0.f;
  }
  // zero rows of the (rare) partial last tile
  if(ebase + MTILE > E){
    for(int p = t; p < MTILE*208; p += 128){
      int s = p/208, f = p%208;
      if(ebase + s >= E) Alds[s*ASTR + f] = (_Float16)0.f;
    }
  }
  asm volatile("s_wait_asynccnt 0x0" ::: "memory");
  __syncthreads();

  // ---- preload B fragments (reused across all 4 M-tiles) ----
  const int wave = t>>5, lane = t&31;
  const int khalf = lane>>4;
  union VH { v16h v; unsigned int u[8]; };
  VH bfr[KT_];
  #pragma unroll
  for(int kt=0; kt<KT_; ++kt)
    bfr[kt].v = *(const v16h*)&wswz[((kt*4 + wave)*32 + lane)*16];

  // ---- WMMA: 4 M-tiles x 7 K-tiles ----
  union VF { v8f v; float f[8]; } accs[4];
  #pragma unroll
  for(int mt=0; mt<4; ++mt){
    #pragma unroll
    for(int r=0;r<8;++r) accs[mt].f[r]=0.f;
    const int rowA = mt*16 + (lane&15);
    #pragma unroll
    for(int kt=0; kt<KT_; ++kt){
      VH a;
      #pragma unroll
      for(int r=0; r<8; ++r){
        int kloc = ((r>=4)?16:0) + khalf*8 + (r&3)*2;
        a.u[r] = *(const unsigned int*)&Alds[rowA*ASTR + kt*32 + kloc];
      }
      accs[mt].v = __builtin_amdgcn_wmma_f32_16x16x32_f16(
          false, a.v, false, bfr[kt].v, (short)0, accs[mt].v, false, false);
    }
  }
  __syncthreads();   // done with A-tile; reuse smem as C staging

  // ---- stage C (bias + ReLU) into LDS ----
  const int colN  = wave*16 + (lane&15);
  const int rbase = (lane>=16)?8:0;
  const float bb = bias[colN];
  #pragma unroll
  for(int mt=0; mt<4; ++mt){
    #pragma unroll
    for(int r=0; r<8; ++r){
      int m = mt*16 + r + rbase;
      float v = accs[mt].f[r] + bb;
      Cst[m*CSTR + colN] = v > 0.f ? v : 0.f;
    }
  }
  __syncthreads();

  // ---- coalesced stores of 64 consecutive edge rows ----
  for(int item = t; item < MTILE*16; item += 128){
    int m = item>>4, q = item&15;
    long e = ebase + m;
    if(e < E){
      float4 v = ((const float4*)&Cst[m*CSTR])[q];
      if(OUTF16){
        union { _Float16 h[4]; uint2 u; } pk;
        pk.h[0]=(_Float16)v.x; pk.h[1]=(_Float16)v.y; pk.h[2]=(_Float16)v.z; pk.h[3]=(_Float16)v.w;
        ((uint2*)ea_out)[e*16 + q] = pk.u;
      } else {
        ((float4*)ea_out)[e*16 + q] = v;
      }
    }
  }
}

__global__ void node_final(const float* agg, const float* cnt, float* out_node, int N){
  long i = blockIdx.x*(long)blockDim.x+threadIdx.x;
  if(i < (long)N*D64){ int n=(int)(i>>6); out_node[i] = agg[i] / (cnt[n]+1.f); }
}

__global__ void graph_pool(const float* node, const int* head, const int* tail,
                           const int* counts, float* out_graph, int B){
  int g = blockIdx.x, c = threadIdx.x;
  if(g >= B || c >= D64) return;
  int start = head[g], n = counts[g];
  float m = -3.402823466e38f;
  for(int i=0;i<n;++i){
    float v = node[(long)(start+i)*D64 + c];
    m = v>m ? v : m;
  }
  out_graph[g*192 + c]       = m;
  out_graph[g*192 + 64 + c]  = node[(long)head[g]*D64 + c];
  out_graph[g*192 + 128 + c] = node[(long)tail[g]*D64 + c];
}

// ---------------- host launcher ----------------
extern "C" void kernel_launch(void* const* d_in, const int* in_sizes, int n_in,
                              void* d_out, int out_size, void* d_ws, size_t ws_size,
                              hipStream_t stream){
  const float* x          = (const float*)d_in[0];
  const int*   edge_index = (const int*)  d_in[1];
  const int*   rel        = (const int*)  d_in[2];
  const int*   batch      = (const int*)  d_in[3];
  const float* mask       = (const float*)d_in[4];
  const float* emb        = (const float*)d_in[5];
  const float* W[3]  = { (const float*)d_in[6], (const float*)d_in[8],  (const float*)d_in[10] };
  const float* bs[3] = { (const float*)d_in[7], (const float*)d_in[9],  (const float*)d_in[11] };

  const int N = in_sizes[0]/NDIM;
  const int E = in_sizes[2];
  const int B = (int)(((long)out_size - (long)N*D64 - (long)E*D64) / 192);

  const int* rowp = edge_index;       // edge_index[0]
  const int* colp = edge_index + E;   // edge_index[1]

  // workspace carve-up
  char* wsb = (char*)d_ws;
  size_t off = 0;
  auto carve = [&](size_t bytes)->void*{ void* p = wsb+off; off += (bytes+255)&~(size_t)255; return p; };
  _Float16* ea16a  = (_Float16*)carve((size_t)E*D64*sizeof(_Float16));
  _Float16* ea16b  = (_Float16*)carve((size_t)E*D64*sizeof(_Float16));
  float*    agg    = (float*)   carve((size_t)N*D64*sizeof(float));
  float*    cnt    = (float*)   carve((size_t)N*sizeof(float));
  _Float16* nrep   = (_Float16*)carve((size_t)N*NSTR*sizeof(_Float16));
  float*    x0     = (float*)   carve((size_t)N*NDIM*sizeof(float));
  _Float16* wswz   = (_Float16*)carve((size_t)3*KT_*4*32*16*sizeof(_Float16));
  int*      counts = (int*)     carve((size_t)B*sizeof(int));
  int*      head   = (int*)     carve((size_t)B*sizeof(int));
  int*      tail   = (int*)     carve((size_t)B*sizeof(int));

  float* out       = (float*)d_out;
  float* out_graph = out;
  float* out_node  = out + (size_t)B*192;
  float* out_ea    = out_node + (size_t)N*D64;

  const dim3 blk(256);
  auto g1 = [&](long n){ return dim3((unsigned)((n+255)/256)); };

  // setup
  zero_i32<<<g1(B), blk, 0, stream>>>(counts, B);
  zero_f32<<<g1(N), blk, 0, stream>>>(cnt, N);
  zero_f32<<<g1((long)N*NDIM), blk, 0, stream>>>(x0, (long)N*NDIM);
  count_batch<<<g1(N), blk, 0, stream>>>(batch, counts, N);
  scan_heads<<<1, 1, 0, stream>>>(counts, head, tail, B);
  scatter_x0<<<g1((long)B*NDIM), blk, 0, stream>>>(x, head, tail, x0, B);
  edge_cnt<<<g1(E), blk, 0, stream>>>(colp, mask, cnt, E);
  gather_emb<<<g1((long)E*16), blk, 0, stream>>>(rel, emb, ea16a, E);
  const int wsz = KT_*4*32*16;
  for(int l=0;l<3;++l)
    prep_w<<<g1(wsz), blk, 0, stream>>>(W[l], wswz + (size_t)l*wsz);

  const int tiles = (E + MTILE - 1) / MTILE;

  // layer 0: ea0(f16, a) -> ea1(f16, b)
  // layer 1: ea1(f16, b) -> ea2(f16, a)
  // layer 2: ea2(f16, a) -> ea3(f32, d_out)
  const _Float16* eins[3]  = { ea16a, ea16b, ea16a };
  _Float16*       eouts16[3]= { ea16b, ea16a, nullptr };
  for(int l=0;l<3;++l){
    zero_f32<<<g1((long)N*D64), blk, 0, stream>>>(agg, (long)N*D64);
    edge_agg_f16<<<g1((long)E*16), blk, 0, stream>>>(colp, mask, eins[l], agg, E);
    node_build<<<g1((long)N*NSTR), blk, 0, stream>>>(agg, cnt, x0, nrep, N);
    if(l < 2){
      edge_gemm<true><<<dim3(tiles), dim3(128), 0, stream>>>(
          rowp, colp, nrep, eins[l], wswz + (size_t)l*wsz, bs[l], (void*)eouts16[l], E);
    } else {
      edge_gemm<false><<<dim3(tiles), dim3(128), 0, stream>>>(
          rowp, colp, nrep, eins[l], wswz + (size_t)l*wsz, bs[l], (void*)out_ea, E);
    }
  }

  // final aggregation (reads the f32 ea in d_out) + pooling
  zero_f32<<<g1((long)N*D64), blk, 0, stream>>>(agg, (long)N*D64);
  edge_agg_f32<<<g1((long)E*16), blk, 0, stream>>>(colp, mask, out_ea, agg, E);
  node_final<<<g1((long)N*D64), blk, 0, stream>>>(agg, cnt, out_node, N);
  graph_pool<<<dim3(B), dim3(64), 0, stream>>>(out_node, head, tail, counts, out_graph, B);
}
